// MultiHeadAttention_61460982005850
// MI455X (gfx1250) — matrix operand order repaired and checked by reading, not compiled, not run on hardware
//
#include <hip/hip_runtime.h>
#include <hip/hip_bf16.h>

// ---------------------------------------------------------------------------
// MultiHeadAttention on gfx1250 (MI455X): bf16 WMMA + async global->LDS copies.
//   B=4, S=2048, D=1024, H=16, hd=64, SCALE = 1/8
// ---------------------------------------------------------------------------

#define D_MODEL   1024
#define SEQ       2048
#define NHEADS    16
#define HDIM      64
#define BATCH     4
#define ROWS_TOT  (BATCH * SEQ)          // 8192
#define ATTN_SCALE 0.125f                // 64^-0.5

typedef __attribute__((ext_vector_type(16))) __bf16 v16bf;
typedef __attribute__((ext_vector_type(8)))  float  v8f;
typedef __attribute__((ext_vector_type(4)))  int    v4i;

struct BF16x16 { uint4 lo, hi; };        // 32 bytes, bit-castable to v16bf

// ---- gfx1250 async global->LDS path (ASYNCcnt), with safe fallback --------
#if defined(__has_builtin)
#if __has_builtin(__builtin_amdgcn_global_load_async_to_lds_b128)
#define ASYNC_LDS 1
#endif
#endif
#ifndef ASYNC_LDS
#define ASYNC_LDS 0
#endif

static __device__ inline void cp_async16(unsigned short* lds_dst,
                                         const unsigned short* gsrc) {
#if ASYNC_LDS
    // signature (probe-confirmed): (v4i AS1*, v4i AS3*, imm int offset, imm int cpol)
    __builtin_amdgcn_global_load_async_to_lds_b128(
        (__attribute__((address_space(1))) v4i*)gsrc,
        (__attribute__((address_space(3))) v4i*)lds_dst, 0, 0);
#else
    *(uint4*)lds_dst = *(const uint4*)gsrc;
#endif
}

static __device__ inline void async_fence() {
#if ASYNC_LDS
#if __has_builtin(__builtin_amdgcn_s_wait_asynccnt)
    __builtin_amdgcn_s_wait_asynccnt(0);
#else
    asm volatile("s_wait_asynccnt 0" ::: "memory");
#endif
#endif
}

// ---------------------------------------------------------------------------
static __device__ inline unsigned short f2bf(float x) {
    unsigned u = __builtin_bit_cast(unsigned, x);
    unsigned r = u + 0x7FFFu + ((u >> 16) & 1u);   // round-to-nearest-even
    return (unsigned short)(r >> 16);
}

static __device__ inline v8f wmma_bf16(v16bf a, v16bf b, v8f c) {
    return __builtin_amdgcn_wmma_f32_16x16x32_bf16(
        false, a, false, b, (short)0, c, false, false);
}

// A-fragment (16x32, M x K) from LDS bf16, row-major, stride in halfs.
// ISA: lane row = lane&15; VGPRs 0..3 hold K = 8*hi+0..7, VGPRs 4..7 hold
// K = 16+8*hi+0..7 (hi = lane>=16).
static __device__ inline v16bf load_a_frag(const unsigned short* lds, int stride) {
    int lane = threadIdx.x & 31;
    int hi   = lane >> 4;
    const unsigned short* p = lds + (lane & 15) * stride;
    BF16x16 t;
    t.lo = *(const uint4*)(p + 8 * hi);
    t.hi = *(const uint4*)(p + 16 + 8 * hi);
    return __builtin_bit_cast(v16bf, t);
}

// B-fragment (32x16, K x N) from LDS stored [N][K]: lane col N = lane&15,
// K = 8*hi+0..7 then 16+8*hi+0..7 (same order as A).
static __device__ inline v16bf load_b_frag(const unsigned short* lds, int stride) {
    int lane = threadIdx.x & 31;
    const unsigned short* p = lds + (lane & 15) * stride;
    BF16x16 t;
    t.lo = *(const uint4*)(p + 8 * (lane >> 4));
    t.hi = *(const uint4*)(p + 16 + 8 * (lane >> 4));
    return __builtin_bit_cast(v16bf, t);
}

static __device__ inline float rmax16(float v) {
    v = fmaxf(v, __shfl_xor(v, 1, 32));
    v = fmaxf(v, __shfl_xor(v, 2, 32));
    v = fmaxf(v, __shfl_xor(v, 4, 32));
    v = fmaxf(v, __shfl_xor(v, 8, 32));
    return v;
}
static __device__ inline float rsum16(float v) {
    v += __shfl_xor(v, 1, 32);
    v += __shfl_xor(v, 2, 32);
    v += __shfl_xor(v, 4, 32);
    v += __shfl_xor(v, 8, 32);
    return v;
}

// ---------------------------------------------------------------------------
// Projection GEMM: dst[b,h,s,d] = bf16( scale * (X @ W)[m,n] )
// X fp32 [8192,1024], W fp32 [1024,1024]. 256 thr (8 waves).
// Tile M=128 (16 rows/wave), N=64, K staged by 32, double-buffered LDS with
// register-level prefetch of the fp32 tiles (they need fp32->bf16 convert).
// ---------------------------------------------------------------------------
__global__ __launch_bounds__(256, 2)
void proj_gemm_kernel(const float* __restrict__ X, const float* __restrict__ W,
                      unsigned short* __restrict__ dst, float scale) {
    __shared__ unsigned short Alds[2][128 * 32];   // [m][k] bf16, 2x8 KB
    __shared__ unsigned short Blds[2][64 * 32];    // [n][k] bf16, 2x4 KB

    const int nbase = blockIdx.x * 64;
    const int mbase = blockIdx.y * 128;
    const int t     = threadIdx.x;
    const int wave  = t >> 5;
    const int lane  = t & 31;
    const int hi    = lane >> 4;
    const int nlo   = lane & 15;

    const int arow = t >> 1, acol = (t & 1) * 16;   // 16 floats/thread of A
    const int brow = t >> 3, bcol = (t & 7) * 8;    // 8 floats/thread of B
    const float* aptr = X + (size_t)(mbase + arow) * D_MODEL + acol;
    const float* bptr = W + (size_t)brow * D_MODEL + nbase + bcol;

    v8f acc[4];
    #pragma unroll
    for (int i = 0; i < 4; ++i) { v8f z = {0,0,0,0,0,0,0,0}; acc[i] = z; }

    float4 af[4], bfv[2];
    #pragma unroll
    for (int j = 0; j < 4; ++j) af[j]  = *(const float4*)(aptr + j * 4);
    #pragma unroll
    for (int j = 0; j < 2; ++j) bfv[j] = *(const float4*)(bptr + j * 4);
    {   // stage tile 0
        unsigned short* da = &Alds[0][arow * 32 + acol];
        #pragma unroll
        for (int j = 0; j < 4; ++j) {
            da[j*4+0] = f2bf(af[j].x); da[j*4+1] = f2bf(af[j].y);
            da[j*4+2] = f2bf(af[j].z); da[j*4+3] = f2bf(af[j].w);
        }
        #pragma unroll
        for (int j = 0; j < 2; ++j) {
            Blds[0][(bcol + j*4 + 0) * 32 + brow] = f2bf(bfv[j].x);
            Blds[0][(bcol + j*4 + 1) * 32 + brow] = f2bf(bfv[j].y);
            Blds[0][(bcol + j*4 + 2) * 32 + brow] = f2bf(bfv[j].z);
            Blds[0][(bcol + j*4 + 3) * 32 + brow] = f2bf(bfv[j].w);
        }
    }
    __syncthreads();

    for (int k0 = 0; k0 < D_MODEL; k0 += 32) {
        const int  cur  = (k0 >> 5) & 1;
        const int  nxt  = cur ^ 1;
        const bool more = (k0 + 32) < D_MODEL;

        if (more) {   // prefetch next fp32 tile into registers (under compute)
            #pragma unroll
            for (int j = 0; j < 4; ++j)
                af[j] = *(const float4*)(aptr + (k0 + 32) + j * 4);
            #pragma unroll
            for (int j = 0; j < 2; ++j)
                bfv[j] = *(const float4*)(bptr + (size_t)(k0 + 32) * D_MODEL + j * 4);
        }

        // ---- compute: hoist all fragments, then 4 back-to-back WMMAs ----
        v16bf a  = load_a_frag(&Alds[cur][wave * 16 * 32], 32);
        v16bf b0 = load_b_frag(&Blds[cur][0 * 16 * 32], 32);
        v16bf b1 = load_b_frag(&Blds[cur][1 * 16 * 32], 32);
        v16bf b2 = load_b_frag(&Blds[cur][2 * 16 * 32], 32);
        v16bf b3 = load_b_frag(&Blds[cur][3 * 16 * 32], 32);
        acc[0] = wmma_bf16(a, b0, acc[0]);
        acc[1] = wmma_bf16(a, b1, acc[1]);
        acc[2] = wmma_bf16(a, b2, acc[2]);
        acc[3] = wmma_bf16(a, b3, acc[3]);

        if (more) {   // convert + stage next tile into the other buffer
            unsigned short* da = &Alds[nxt][arow * 32 + acol];
            #pragma unroll
            for (int j = 0; j < 4; ++j) {
                da[j*4+0] = f2bf(af[j].x); da[j*4+1] = f2bf(af[j].y);
                da[j*4+2] = f2bf(af[j].z); da[j*4+3] = f2bf(af[j].w);
            }
            #pragma unroll
            for (int j = 0; j < 2; ++j) {
                Blds[nxt][(bcol + j*4 + 0) * 32 + brow] = f2bf(bfv[j].x);
                Blds[nxt][(bcol + j*4 + 1) * 32 + brow] = f2bf(bfv[j].y);
                Blds[nxt][(bcol + j*4 + 2) * 32 + brow] = f2bf(bfv[j].z);
                Blds[nxt][(bcol + j*4 + 3) * 32 + brow] = f2bf(bfv[j].w);
            }
        }
        __syncthreads();
    }

    // Store bf16 with head permutation: [m=b*S+s, n=h*64+d] -> [b,h,s,d]
    #pragma unroll
    for (int nt = 0; nt < 4; ++nt) {
        #pragma unroll
        for (int r = 0; r < 8; ++r) {
            int m = mbase + wave * 16 + r + 8 * hi;
            int n = nbase + nt * 16 + nlo;
            int b = m >> 11, s = m & (SEQ - 1);
            int h = n >> 6,  d = n & (HDIM - 1);
            size_t idx = (((size_t)(b * NHEADS + h) * SEQ + s) * HDIM) + d;
            dst[idx] = f2bf(acc[nt][r] * scale);
        }
    }
}

// ---------------------------------------------------------------------------
// Output GEMM: out fp32 = ctx(bf16 [8192,1024]) @ Wo(fp32 [1024,1024]).
// A tile is already bf16 -> streamed with async global->LDS copies.
// ---------------------------------------------------------------------------
__global__ __launch_bounds__(256, 2)
void out_gemm_kernel(const unsigned short* __restrict__ A,
                     const float* __restrict__ W, float* __restrict__ out) {
    __shared__ unsigned short Alds[2][128 * 32];
    __shared__ unsigned short Blds[2][64 * 32];

    const int nbase = blockIdx.x * 64;
    const int mbase = blockIdx.y * 128;
    const int t     = threadIdx.x;
    const int wave  = t >> 5;
    const int lane  = t & 31;
    const int hi    = lane >> 4;
    const int nlo   = lane & 15;

    const int arow = t >> 1, acol = (t & 1) * 16;   // 16 halfs (32B) of A
    const int brow = t >> 3, bcol = (t & 7) * 8;
    const unsigned short* aptr = A + (size_t)(mbase + arow) * D_MODEL + acol;
    const float* bptr = W + (size_t)brow * D_MODEL + nbase + bcol;

    v8f acc[4];
    #pragma unroll
    for (int i = 0; i < 4; ++i) { v8f z = {0,0,0,0,0,0,0,0}; acc[i] = z; }

    // stage tile 0: A via async copy, B via convert
    cp_async16(&Alds[0][arow * 32 + acol], aptr);
    cp_async16(&Alds[0][arow * 32 + acol + 8], aptr + 8);
    float4 bfv[2];
    #pragma unroll
    for (int j = 0; j < 2; ++j) bfv[j] = *(const float4*)(bptr + j * 4);
    #pragma unroll
    for (int j = 0; j < 2; ++j) {
        Blds[0][(bcol + j*4 + 0) * 32 + brow] = f2bf(bfv[j].x);
        Blds[0][(bcol + j*4 + 1) * 32 + brow] = f2bf(bfv[j].y);
        Blds[0][(bcol + j*4 + 2) * 32 + brow] = f2bf(bfv[j].z);
        Blds[0][(bcol + j*4 + 3) * 32 + brow] = f2bf(bfv[j].w);
    }
    async_fence();
    __syncthreads();

    for (int k0 = 0; k0 < D_MODEL; k0 += 32) {
        const int  cur  = (k0 >> 5) & 1;
        const int  nxt  = cur ^ 1;
        const bool more = (k0 + 32) < D_MODEL;

        if (more) {   // async-prefetch next A tile straight into LDS
            cp_async16(&Alds[nxt][arow * 32 + acol], aptr + k0 + 32);
            cp_async16(&Alds[nxt][arow * 32 + acol + 8], aptr + k0 + 32 + 8);
            #pragma unroll
            for (int j = 0; j < 2; ++j)
                bfv[j] = *(const float4*)(bptr + (size_t)(k0 + 32) * D_MODEL + j * 4);
        }

        v16bf a  = load_a_frag(&Alds[cur][wave * 16 * 32], 32);
        v16bf b0 = load_b_frag(&Blds[cur][0 * 16 * 32], 32);
        v16bf b1 = load_b_frag(&Blds[cur][1 * 16 * 32], 32);
        v16bf b2 = load_b_frag(&Blds[cur][2 * 16 * 32], 32);
        v16bf b3 = load_b_frag(&Blds[cur][3 * 16 * 32], 32);
        acc[0] = wmma_bf16(a, b0, acc[0]);
        acc[1] = wmma_bf16(a, b1, acc[1]);
        acc[2] = wmma_bf16(a, b2, acc[2]);
        acc[3] = wmma_bf16(a, b3, acc[3]);

        if (more) {
            #pragma unroll
            for (int j = 0; j < 2; ++j) {
                Blds[nxt][(bcol + j*4 + 0) * 32 + brow] = f2bf(bfv[j].x);
                Blds[nxt][(bcol + j*4 + 1) * 32 + brow] = f2bf(bfv[j].y);
                Blds[nxt][(bcol + j*4 + 2) * 32 + brow] = f2bf(bfv[j].z);
                Blds[nxt][(bcol + j*4 + 3) * 32 + brow] = f2bf(bfv[j].w);
            }
        }
        async_fence();
        __syncthreads();
    }

    #pragma unroll
    for (int nt = 0; nt < 4; ++nt)
        #pragma unroll
        for (int r = 0; r < 8; ++r) {
            int m = mbase + wave * 16 + r + 8 * hi;
            int n = nbase + nt * 16 + nlo;
            out[(size_t)m * D_MODEL + n] = acc[nt][r];
        }
}

// ---------------------------------------------------------------------------
// Flash attention: Q,K,V bf16 in [B,H,S,64]; ctx bf16 out in [B,S,H*64].
// 128 thr (4 waves), 64 q rows/block, 64-key KV tiles. K tiles streamed with
// async global->LDS double buffering; V prefetched in registers (needs LDS
// transpose for the PV B-fragments); online softmax with wave32 shfl_xor
// reductions inside each 16-lane N-group.
// ---------------------------------------------------------------------------
__global__ __launch_bounds__(128, 2)
void attn_kernel(const unsigned short* __restrict__ Qh,
                 const unsigned short* __restrict__ Kh,
                 const unsigned short* __restrict__ Vh,
                 const float* __restrict__ masks,
                 unsigned short* __restrict__ ctx) {
    __shared__ unsigned short Qlds[64 * 64];          // [q][d]    8 KB
    __shared__ unsigned short Klds[2][64 * 64];       // [key][d]  16 KB
    __shared__ unsigned short Vtlds[2][64 * 64];      // [d][key]  16 KB
    __shared__ unsigned short Plds[4][16 * 64];       // per-wave probs, 8 KB

    const int bh    = blockIdx.y;                     // 0..63
    const int b     = bh >> 4;
    const int h     = bh & 15;
    const int qbase = blockIdx.x * 64;
    const size_t headoff = (size_t)bh * SEQ * HDIM;
    const unsigned short* Q = Qh + headoff;
    const unsigned short* K = Kh + headoff;
    const unsigned short* V = Vh + headoff;

    const int t    = threadIdx.x;
    const int wave = t >> 5;
    const int lane = t & 31;
    const int hi   = lane >> 4;
    const int nlo  = lane & 15;
    const int row  = t >> 1;                          // 0..63 copy row
    const int col  = (t & 1) * 32;                    // 0 / 32 halfs

    {   // Stage Q tile (64x64 bf16) via async copies
        const unsigned short* qs = Q + (size_t)(qbase + row) * HDIM + col;
        #pragma unroll
        for (int j = 0; j < 4; ++j)
            cp_async16(&Qlds[row * 64 + col + j * 8], qs + j * 8);
    }
    {   // Prologue: K tile 0 async, V tile 0 via registers + transpose store
        const unsigned short* ks = K + (size_t)row * HDIM + col;
        #pragma unroll
        for (int j = 0; j < 4; ++j)
            cp_async16(&Klds[0][row * 64 + col + j * 8], ks + j * 8);
        uint4 vreg[4];
        const unsigned short* vs = V + (size_t)row * HDIM + col;
        #pragma unroll
        for (int j = 0; j < 4; ++j) vreg[j] = *(const uint4*)(vs + j * 8);
        const unsigned short* hh = (const unsigned short*)vreg;
        #pragma unroll
        for (int j = 0; j < 32; ++j)
            Vtlds[0][(col + j) * 64 + row] = hh[j];
    }
    async_fence();
    __syncthreads();

    const v16bf qa0 = load_a_frag(&Qlds[wave * 16 * 64], 64);       // dims 0..31
    const v16bf qa1 = load_a_frag(&Qlds[wave * 16 * 64 + 32], 64);  // dims 32..63

    float mstate[8], lstate[8];
    v8f oacc[4];
    #pragma unroll
    for (int r = 0; r < 8; ++r) { mstate[r] = -1e30f; lstate[r] = 0.f; }
    #pragma unroll
    for (int i = 0; i < 4; ++i) { v8f z = {0,0,0,0,0,0,0,0}; oacc[i] = z; }

    for (int it = 0; it < SEQ / 64; ++it) {
        const int  cur  = it & 1;
        const int  nxt  = cur ^ 1;
        const int  kv   = it * 64;
        const bool more = (it + 1) < (SEQ / 64);

        uint4 vreg[4];
        if (more) {   // prefetch tile it+1: K async into LDS, V into registers
            const unsigned short* ks = K + ((size_t)(kv + 64) + row) * HDIM + col;
            #pragma unroll
            for (int j = 0; j < 4; ++j)
                cp_async16(&Klds[nxt][row * 64 + col + j * 8], ks + j * 8);
            const unsigned short* vs = V + ((size_t)(kv + 64) + row) * HDIM + col;
            #pragma unroll
            for (int j = 0; j < 4; ++j) vreg[j] = *(const uint4*)(vs + j * 8);
        }

        // ---- logits = Q @ K^T : hoist fragments, 4 WMMAs per k-chunk ----
        v8f sacc[4];
        #pragma unroll
        for (int i = 0; i < 4; ++i) { v8f z = {0,0,0,0,0,0,0,0}; sacc[i] = z; }
        #pragma unroll
        for (int kc = 0; kc < 2; ++kc) {
            v16bf aq = kc ? qa1 : qa0;
            v16bf b0 = load_b_frag(&Klds[cur][0 * 16 * 64 + kc * 32], 64);
            v16bf b1 = load_b_frag(&Klds[cur][1 * 16 * 64 + kc * 32], 64);
            v16bf b2 = load_b_frag(&Klds[cur][2 * 16 * 64 + kc * 32], 64);
            v16bf b3 = load_b_frag(&Klds[cur][3 * 16 * 64 + kc * 32], 64);
            sacc[0] = wmma_bf16(aq, b0, sacc[0]);
            sacc[1] = wmma_bf16(aq, b1, sacc[1]);
            sacc[2] = wmma_bf16(aq, b2, sacc[2]);
            sacc[3] = wmma_bf16(aq, b3, sacc[3]);
        }
        // additive mask (broadcast over q rows)
        #pragma unroll
        for (int nt = 0; nt < 4; ++nt) {
            float mk = masks[(size_t)b * SEQ + kv + nt * 16 + nlo];
            #pragma unroll
            for (int r = 0; r < 8; ++r) sacc[nt][r] += mk;
        }

        // ---- online softmax (lane's row for vgpr r is r + 8*hi) ----
        #pragma unroll
        for (int r = 0; r < 8; ++r) {
            float tv = fmaxf(fmaxf(sacc[0][r], sacc[1][r]),
                             fmaxf(sacc[2][r], sacc[3][r]));
            float mnew = fmaxf(mstate[r], rmax16(tv));
            float rescale = __expf(mstate[r] - mnew);
            float psum = 0.f;
            #pragma unroll
            for (int nt = 0; nt < 4; ++nt) {
                float p = __expf(sacc[nt][r] - mnew);
                psum += p;
                Plds[wave][(r + 8 * hi) * 64 + nt * 16 + nlo] = f2bf(p);
            }
            lstate[r] = lstate[r] * rescale + rsum16(psum);
            mstate[r] = mnew;
            #pragma unroll
            for (int nt = 0; nt < 4; ++nt) oacc[nt][r] *= rescale;
        }

        // ---- O += P @ V (Plds region private per wave; DS ops in-order) ----
        #pragma unroll
        for (int kc = 0; kc < 2; ++kc) {
            v16bf pa = load_a_frag(&Plds[wave][kc * 32], 64);
            v16bf v0 = load_b_frag(&Vtlds[cur][0 * 16 * 64 + kc * 32], 64);
            v16bf v1 = load_b_frag(&Vtlds[cur][1 * 16 * 64 + kc * 32], 64);
            v16bf v2 = load_b_frag(&Vtlds[cur][2 * 16 * 64 + kc * 32], 64);
            v16bf v3 = load_b_frag(&Vtlds[cur][3 * 16 * 64 + kc * 32], 64);
            oacc[0] = wmma_bf16(pa, v0, oacc[0]);
            oacc[1] = wmma_bf16(pa, v1, oacc[1]);
            oacc[2] = wmma_bf16(pa, v2, oacc[2]);
            oacc[3] = wmma_bf16(pa, v3, oacc[3]);
        }

        if (more) {   // stage prefetched V (transposed) into the other buffer
            const unsigned short* hh = (const unsigned short*)vreg;
            #pragma unroll
            for (int j = 0; j < 32; ++j)
                Vtlds[nxt][(col + j) * 64 + row] = hh[j];
        }
        async_fence();     // K tile (it+1) resident before the barrier
        __syncthreads();
    }

    // Epilogue: normalize + store ctx bf16 as [b, s, h*64 + d]
    #pragma unroll
    for (int r = 0; r < 8; ++r) {
        float inv = 1.0f / lstate[r];
        int s = qbase + wave * 16 + r + 8 * hi;
        #pragma unroll
        for (int nt = 0; nt < 4; ++nt) {
            int d = nt * 16 + nlo;
            size_t idx = ((size_t)(b * SEQ + s) * D_MODEL) + h * HDIM + d;
            ctx[idx] = f2bf(oacc[nt][r] * inv);
        }
    }
}

// ---------------------------------------------------------------------------
extern "C" void kernel_launch(void* const* d_in, const int* in_sizes, int n_in,
                              void* d_out, int out_size, void* d_ws, size_t ws_size,
                              hipStream_t stream) {
    const float* q_in  = (const float*)d_in[0];
    const float* k_in  = (const float*)d_in[1];
    const float* v_in  = (const float*)d_in[2];
    const float* masks = (const float*)d_in[3];
    const float* Wq    = (const float*)d_in[4];
    const float* Wk    = (const float*)d_in[5];
    const float* Wv    = (const float*)d_in[6];
    const float* Wo    = (const float*)d_in[7];

    const size_t elems = (size_t)ROWS_TOT * D_MODEL;   // 8M bf16 = 16 MB each
    unsigned short* Qh  = (unsigned short*)d_ws;
    unsigned short* Kh  = Qh + elems;
    unsigned short* Vh  = Kh + elems;
    unsigned short* Ctx = Vh + elems;

    dim3 gemm_grid(D_MODEL / 64, ROWS_TOT / 128);      // (16, 64)
    dim3 gemm_blk(256);
    proj_gemm_kernel<<<gemm_grid, gemm_blk, 0, stream>>>(q_in, Wq, Qh, ATTN_SCALE);
    proj_gemm_kernel<<<gemm_grid, gemm_blk, 0, stream>>>(k_in, Wk, Kh, 1.0f);
    proj_gemm_kernel<<<gemm_grid, gemm_blk, 0, stream>>>(v_in, Wv, Vh, 1.0f);

    dim3 attn_grid(SEQ / 64, BATCH * NHEADS);          // (32, 64)
    attn_kernel<<<attn_grid, dim3(128), 0, stream>>>(Qh, Kh, Vh, masks, Ctx);

    out_gemm_kernel<<<gemm_grid, gemm_blk, 0, stream>>>(Ctx, Wo, (float*)d_out);
}
